// SingleLayerGCN_25692494364820
// MI455X (gfx1250) — matrix operand order repaired and checked by reading, not compiled, not run on hardware
//
#include <hip/hip_runtime.h>
#include <hip/hip_bf16.h>
#include <stdint.h>

// ---------------- problem constants ----------------
#define NNODES 100000
#define INCH   512
#define OUTCH  64
#define NEDGES 1600000
#define MTILES (NNODES / 16)       // 6250 exact
#define KSTEPS (INCH / 32)         // 16
#define NTILES (OUTCH / 16)        // 4
#define WSWZ_DWORDS (NTILES * KSTEPS * 32 * 8)   // 16384 dwords = 64KB

typedef __attribute__((ext_vector_type(16))) __bf16 v16bf;
typedef __attribute__((ext_vector_type(16))) float  v16f;
typedef __attribute__((ext_vector_type(8)))  float  v8f;

union BFrag { uint4 q[2]; uint32_t u[8]; v16bf v; };

// round-to-nearest-even f32 -> bf16, pack two into one dword (lo=a, hi=b)
// (used only in the tiny one-shot W swizzle kernel)
__device__ __forceinline__ uint32_t pk_bf16(float a, float b) {
  uint32_t ua = __float_as_uint(a);
  uint32_t ub = __float_as_uint(b);
  ua += 0x7FFFu + ((ua >> 16) & 1u);
  ub += 0x7FFFu + ((ub >> 16) & 1u);
  return (ua >> 16) | (ub & 0xFFFF0000u);
}

// ---------------- degree / normalization ----------------
__global__ void k_init_deg(float* deg) {
  int i = blockIdx.x * blockDim.x + threadIdx.x;
  if (i < NNODES) deg[i] = 1.0f;              // self-loop contributes 1
}

__global__ void k_count_deg(const int* __restrict__ ei, float* deg) {
  int e = blockIdx.x * blockDim.x + threadIdx.x;
  if (e < NEDGES) atomicAdd(&deg[ei[NEDGES + e]], 1.0f);
}

__global__ void k_rsqrt_deg(float* deg) {     // in place: deg -> deg^{-1/2}
  int i = blockIdx.x * blockDim.x + threadIdx.x;
  if (i < NNODES) deg[i] = rsqrtf(deg[i]);
}

// ---------------- W pre-swizzle into WMMA-B bf16 fragment layout ----------------
// entry q = t*4096 + s*256 + lane*8 + j ; lane<16: N=t*16+lane, K=s*32+2j(+1)
//                                        lane>=16: N=t*16+lane-16, K=s*32+16+2j(+1)
__global__ void k_swizzle_w(const float* __restrict__ W, uint32_t* __restrict__ wswz) {
  int q = blockIdx.x * blockDim.x + threadIdx.x;
  if (q >= WSWZ_DWORDS) return;
  int j    = q & 7;
  int lane = (q >> 3) & 31;
  int s    = (q >> 8) & 15;
  int t    = q >> 12;
  int n, kb;
  if (lane < 16) { n = t * 16 + lane;      kb = s * 32 + 2 * j; }
  else           { n = t * 16 + lane - 16; kb = s * 32 + 16 + 2 * j; }
  float w0 = W[kb * OUTCH + n];
  float w1 = W[(kb + 1) * OUTCH + n];
  wswz[q] = pk_bf16(w0, w1);
}

// ---------------- GEMM: h = x @ W (bf16 WMMA, f32 acc) + self-loop agg init ----------------
__global__ void __launch_bounds__(256)
k_gemm(const float* __restrict__ x, const uint32_t* __restrict__ wswz,
       const float* __restrict__ dinv, float* __restrict__ h,
       float* __restrict__ agg) {
  __shared__ uint32_t ldsB[WSWZ_DWORDS];      // 64 KB

  // cooperative coalesced copy of pre-swizzled W into LDS
  {
    const uint4* src = (const uint4*)wswz;
    uint4* dst = (uint4*)ldsB;
    for (int q = threadIdx.x; q < WSWZ_DWORDS / 4; q += 256) dst[q] = src[q];
  }
  __syncthreads();

  const int wave = threadIdx.x >> 5;
  const int lane = threadIdx.x & 31;
  const int tile = blockIdx.x * 8 + wave;     // 16-row M tile
  if (tile >= MTILES) return;

  const int m0      = tile * 16;
  const int laneHi  = lane >> 4;              // 0: lanes 0-15, 1: lanes 16-31
  const int laneLo  = lane & 15;
  const int row     = m0 + laneLo;            // A-matrix row for this lane
  const float* xrow = x + (size_t)row * INCH;

  v8f zero = {0.f, 0.f, 0.f, 0.f, 0.f, 0.f, 0.f, 0.f};
  v8f acc[NTILES];
#pragma unroll
  for (int t = 0; t < NTILES; ++t) acc[t] = zero;

  const uint4* ldsB4 = (const uint4*)ldsB;

  for (int s = 0; s < KSTEPS; ++s) {
    const int k0   = s * 32;
    const int base = k0 + (laneHi ? 16 : 0);
    // one contiguous 16-float run: K = base..base+15
    float4 f0 = *(const float4*)(xrow + base);
    float4 f1 = *(const float4*)(xrow + base + 4);
    float4 f2 = *(const float4*)(xrow + base + 8);
    float4 f3 = *(const float4*)(xrow + base + 12);

    // native pk converts (RTNE) instead of manual bit-twiddling
    v16f vf;
    vf[0]  = f0.x; vf[1]  = f0.y; vf[2]  = f0.z; vf[3]  = f0.w;
    vf[4]  = f1.x; vf[5]  = f1.y; vf[6]  = f1.z; vf[7]  = f1.w;
    vf[8]  = f2.x; vf[9]  = f2.y; vf[10] = f2.z; vf[11] = f2.w;
    vf[12] = f3.x; vf[13] = f3.y; vf[14] = f3.z; vf[15] = f3.w;
    v16bf av = __builtin_convertvector(vf, v16bf);

#pragma unroll
    for (int t = 0; t < NTILES; ++t) {
      BFrag b;
      const int idx4 = ((t * KSTEPS + s) * 32 + lane) * 2;
      b.q[0] = ldsB4[idx4];
      b.q[1] = ldsB4[idx4 + 1];
      acc[t] = __builtin_amdgcn_wmma_f32_16x16x32_bf16(
          false, av, false, b.v, (short)0, acc[t], false, false);
    }
  }

  // epilogue: C layout VGPR r -> M = m0 + r + 8*laneHi ; N = t*16 + laneLo
#pragma unroll
  for (int r = 0; r < 8; ++r) {
    const int m  = m0 + r + (laneHi ? 8 : 0);
    const float dv  = dinv[m];
    const float dv2 = dv * dv;
    float* hrow = h   + (size_t)m * OUTCH;
    float* arow = agg + (size_t)m * OUTCH;
#pragma unroll
    for (int t = 0; t < NTILES; ++t) {
      const int n   = t * 16 + laneLo;
      const float v = acc[t][r];
      hrow[n] = v;
      arow[n] = v * dv2;    // self-loop message (norm = dinv^2)
    }
  }
}

// ---------------- edge scatter: agg[col] += h[row] * dinv[row]*dinv[col] ----------------
__global__ void k_edge(const int* __restrict__ ei, const float* __restrict__ h,
                       const float* __restrict__ dinv, float* __restrict__ agg) {
  int gid  = blockIdx.x * blockDim.x + threadIdx.x;
  int e    = gid >> 5;
  int lane = gid & 31;
  if (e >= NEDGES) return;
  int row = ei[e];
  int col = ei[NEDGES + e];
  float norm = dinv[row] * dinv[col];
  const float* hr = h + (size_t)row * OUTCH;
  float* ar = agg + (size_t)col * OUTCH;
  float v0 = hr[lane]      * norm;
  float v1 = hr[lane + 32] * norm;
  atomicAdd(&ar[lane],      v0);
  atomicAdd(&ar[lane + 32], v1);
}

// ---------------- out = log_softmax(agg + b), one wave per node, in place ----------------
__global__ void k_logsoftmax(float* __restrict__ out, const float* __restrict__ bias) {
  int gid  = blockIdx.x * blockDim.x + threadIdx.x;
  int node = gid >> 5;
  int lane = gid & 31;
  if (node >= NNODES) return;
  float* row = out + (size_t)node * OUTCH;
  float v0 = row[lane]      + bias[lane];
  float v1 = row[lane + 32] + bias[lane + 32];
  float m = fmaxf(v0, v1);
#pragma unroll
  for (int off = 16; off > 0; off >>= 1) m = fmaxf(m, __shfl_xor(m, off, 32));
  float s = __expf(v0 - m) + __expf(v1 - m);
#pragma unroll
  for (int off = 16; off > 0; off >>= 1) s += __shfl_xor(s, off, 32);
  float ls = __logf(s);
  row[lane]      = v0 - m - ls;
  row[lane + 32] = v1 - m - ls;
}

// ---------------- launch ----------------
extern "C" void kernel_launch(void* const* d_in, const int* in_sizes, int n_in,
                              void* d_out, int out_size, void* d_ws, size_t ws_size,
                              hipStream_t stream) {
  const float* x  = (const float*)d_in[0];
  const int*   ei = (const int*)d_in[1];
  const float* W  = (const float*)d_in[2];
  const float* b  = (const float*)d_in[3];
  float* out = (float*)d_out;

  // workspace layout (floats): [0,100000) deg/dinv ; [131072,+16384) swizzled W ;
  // [147456, +6400000) h
  float*    ws   = (float*)d_ws;
  float*    deg  = ws;
  uint32_t* wswz = (uint32_t*)(ws + 131072);
  float*    h    = ws + 147456;

  k_init_deg <<<(NNODES + 255) / 256, 256, 0, stream>>>(deg);
  k_count_deg<<<(NEDGES + 255) / 256, 256, 0, stream>>>(ei, deg);
  k_rsqrt_deg<<<(NNODES + 255) / 256, 256, 0, stream>>>(deg);
  k_swizzle_w<<<WSWZ_DWORDS / 256, 256, 0, stream>>>(W, wswz);
  k_gemm     <<<(MTILES + 7) / 8, 256, 0, stream>>>(x, wswz, deg, h, out);
  k_edge     <<<(NEDGES * 32) / 256, 256, 0, stream>>>(ei, h, deg, out);
  k_logsoftmax<<<(NNODES * 32) / 256, 256, 0, stream>>>(out, b);
}
